// CustomLstm_88502096101611
// MI455X (gfx1250) — matrix operand order repaired and checked by reading, not compiled, not run on hardware
//
#include <hip/hip_runtime.h>
#include <hip/hip_bf16.h>

#define D 2048
#define H 1024
#define LDP 40   // padded LDS stride in bf16 elems (80B = 20 banks -> conflict-free b128)

typedef __attribute__((ext_vector_type(16))) __bf16 v16bf;
typedef __attribute__((ext_vector_type(8)))  __bf16 v8bf;
typedef __attribute__((ext_vector_type(8)))  float  v8f;
typedef __attribute__((ext_vector_type(4)))  float  v4f;

union FragU { v16bf v; v8bf h[2]; };

__device__ __forceinline__ float fast_sigmoid(float x) {
    return 1.0f / (1.0f + __expf(-x));
}
__device__ __forceinline__ float fast_tanh(float x) {
    // tanh(x) = 1 - 2/(e^{2x}+1); saturates correctly at +-1
    return 1.0f - 2.0f / (__expf(2.0f * x) + 1.0f);
}

// convert two f32x4 registers to 8 bf16 (v_cvt_pk_bf16_f32 x4)
__device__ __forceinline__ v8bf cvt8r(v4f a, v4f b) {
    v8bf r;
    r[0] = (__bf16)a[0]; r[1] = (__bf16)a[1]; r[2] = (__bf16)a[2]; r[3] = (__bf16)a[3];
    r[4] = (__bf16)b[0]; r[5] = (__bf16)b[1]; r[6] = (__bf16)b[2]; r[7] = (__bf16)b[3];
    return r;
}

// A fragment (16x32 bf16, row-major LDS tile): lane<16 -> K{0..7,16..23}, lane>=16 -> K{8..15,24..31}
__device__ __forceinline__ v16bf load_a_frag(const __bf16* tile, int row, int lhi) {
    FragU u;
    u.h[0] = *(const v8bf*)(tile + row * LDP + lhi * 8);
    u.h[1] = *(const v8bf*)(tile + row * LDP + lhi * 8 + 16);
    return u.v;
}
// B fragment (32x16 bf16, column-major LDS tile): lane = column, lane<16 -> K{0..7,16..23}, lane>=16 -> K{8..15,24..31}
__device__ __forceinline__ v16bf load_b_frag(const __bf16* tile, int col, int lhi) {
    FragU u;
    u.h[0] = *(const v8bf*)(tile + col * LDP + lhi * 8);
    u.h[1] = *(const v8bf*)(tile + col * LDP + lhi * 8 + 16);
    return u.v;
}

// ---------------------------------------------------------------------------
// Kernel 1: fused 4-gate GEMM (w1..w4 @ [hPrev;xt]) + bias + gates + ct/ht
// 256 threads (8 waves) compute a 64x64 tile of ALL 4 gate pre-activations.
// LDS double-buffered, one barrier per K-step. Staging issues all 10 global
// b128 loads in one clause before any convert/store, so the whole tile costs
// one memory round trip that overlaps the WMMA burst.
// ---------------------------------------------------------------------------
__global__ __launch_bounds__(256)
void lstm_gates_kernel(const float* __restrict__ cPrev,
                       const float* __restrict__ hPrev,
                       const float* __restrict__ xt,
                       const float* __restrict__ w1, const float* __restrict__ b1,
                       const float* __restrict__ w2, const float* __restrict__ b2,
                       const float* __restrict__ w3, const float* __restrict__ b3,
                       const float* __restrict__ w4, const float* __restrict__ b4,
                       float* __restrict__ ct_out, float* __restrict__ ht_out)
{
    __shared__ __bf16 As[2][4][64][LDP];   // double-buffered A tiles (per gate), row-major
    __shared__ __bf16 Bs[2][64][LDP];      // double-buffered B tile, column-major: Bs[.][n][k]

    const int tid   = threadIdx.x;
    const int lane  = tid & 31;
    const int wid   = tid >> 5;
    const int waveM = wid >> 1;      // 0..3  (rows)
    const int waveN = wid & 1;       // 0..1  (cols)
    const int lr    = lane & 15;
    const int lhi   = lane >> 4;

    const int tileM = blockIdx.y * 64;
    const int tileN = blockIdx.x * 64;

    const float* W[4] = {w1, w2, w3, w4};
    const float* B[4] = {b1, b2, b3, b4};

    // staging decomposition: A: 64 rows x 4 chunks of 8 (per gate); B: 32 k x 8 chunks of 8
    const int aRow = tid >> 2;          // 0..63
    const int aK8  = (tid & 3) * 8;     // 0,8,16,24
    const int bK   = tid >> 3;          // 0..31
    const int bN8  = (tid & 7) * 8;     // 0..56

    v8f acc[4][2];
    #pragma unroll
    for (int g = 0; g < 4; ++g)
        #pragma unroll
        for (int n = 0; n < 2; ++n)
            #pragma unroll
            for (int e = 0; e < 8; ++e) acc[g][n][e] = 0.0f;

    auto stage = [&](int buf, int kt) {
        // phase 1: issue ALL global loads (one clause, one round trip)
        v4f ta[4][2];
        v4f tb[2];
        #pragma unroll
        for (int g = 0; g < 4; ++g) {
            const float* src = W[g] + (size_t)(tileM + aRow) * D + kt + aK8;
            ta[g][0] = *(const v4f*)(src);
            ta[g][1] = *(const v4f*)(src + 4);
        }
        const int kk = kt + bK;
        const float* srcB = (kk < H) ? (hPrev + (size_t)kk * D + tileN + bN8)
                                     : (xt + (size_t)(kk - H) * D + tileN + bN8);
        tb[0] = *(const v4f*)(srcB);
        tb[1] = *(const v4f*)(srcB + 4);

        // phase 2: convert + LDS store as loads drain
        #pragma unroll
        for (int g = 0; g < 4; ++g)
            *(v8bf*)&As[buf][g][aRow][aK8] = cvt8r(ta[g][0], ta[g][1]);
        v8bf bv = cvt8r(tb[0], tb[1]);
        #pragma unroll
        for (int j = 0; j < 8; ++j) Bs[buf][bN8 + j][bK] = bv[j];
    };

    auto compute = [&](int buf) {
        v16bf aFrag[4];
        #pragma unroll
        for (int g = 0; g < 4; ++g)
            aFrag[g] = load_a_frag(&As[buf][g][0][0], waveM * 16 + lr, lhi);
        v16bf bFrag[2];
        #pragma unroll
        for (int n = 0; n < 2; ++n)
            bFrag[n] = load_b_frag(&Bs[buf][0][0], waveN * 32 + n * 16 + lr, lhi);
        #pragma unroll
        for (int g = 0; g < 4; ++g)
            #pragma unroll
            for (int n = 0; n < 2; ++n)
                acc[g][n] = __builtin_amdgcn_wmma_f32_16x16x32_bf16(
                    false, aFrag[g], false, bFrag[n], (short)0, acc[g][n], false, false);
    };

    // epilogue element coordinates (C layout: VGPR r -> M = r + lhi*8, N = lr)
    const int rowBase = tileM + waveM * 16 + lhi * 8;
    const int colBase = tileN + waveN * 32 + lr;

    stage(0, 0);
    int buf = 0;
    for (int kt = 0; kt < D; kt += 32) {
        __syncthreads();
        compute(buf);                         // 8 WMMAs into the matrix pipe
        if (kt + 32 < D) {
            stage(buf ^ 1, kt + 32);          // overlap next-tile staging with WMMAs
        } else {
            // warm L2/L0 for the scattered epilogue operands (global_prefetch_b8)
            const size_t pidx = (size_t)rowBase * D + colBase;
            __builtin_prefetch(cPrev + pidx, 0, 1);
            __builtin_prefetch(B[0] + pidx, 0, 1);
            __builtin_prefetch(B[1] + pidx, 0, 1);
            __builtin_prefetch(B[2] + pidx, 0, 1);
            __builtin_prefetch(B[3] + pidx, 0, 1);
        }
        buf ^= 1;
    }

    // ---- epilogue: bias + gates + ct/ht ----
    #pragma unroll
    for (int n = 0; n < 2; ++n) {
        const int col = colBase + n * 16;
        #pragma unroll
        for (int r = 0; r < 8; ++r) {
            const size_t idx = (size_t)(rowBase + r) * D + col;
            const float p1 = acc[0][n][r] + B[0][idx];
            const float p2 = acc[1][n][r] + B[1][idx];
            const float p3 = acc[2][n][r] + B[2][idx];
            const float p4 = acc[3][n][r] + B[3][idx];
            const float fg = fast_sigmoid(p1);   // forget
            const float ug = fast_sigmoid(p2);   // update
            const float gg = fast_tanh(p3);      // candidate_gate
            const float cg = fast_sigmoid(p4);   // candidate
            const float c  = fg * cPrev[idx] + ug * gg;
            const float h  = cg * fast_tanh(c);
            ct_out[idx] = c;
            ht_out[idx] = h;
        }
    }
}

// ---------------------------------------------------------------------------
// Kernel 2: logits = w5 @ ht + b5  (ht read fp32 from d_out, cvt bf16 on the fly)
// Same double-buffered, load-all-then-convert skeleton, single matrix.
// ---------------------------------------------------------------------------
__global__ __launch_bounds__(256)
void lstm_out_gemm_kernel(const float* __restrict__ w5, const float* __restrict__ b5,
                          const float* __restrict__ ht, float* __restrict__ logits)
{
    __shared__ __bf16 As[2][64][LDP];
    __shared__ __bf16 Bs[2][64][LDP];

    const int tid   = threadIdx.x;
    const int lane  = tid & 31;
    const int wid   = tid >> 5;
    const int waveM = wid >> 1;
    const int waveN = wid & 1;
    const int lr    = lane & 15;
    const int lhi   = lane >> 4;

    const int tileM = blockIdx.y * 64;
    const int tileN = blockIdx.x * 64;

    const int aRow = tid >> 2;
    const int aK8  = (tid & 3) * 8;
    const int bK   = tid >> 3;
    const int bN8  = (tid & 7) * 8;

    v8f acc[2];
    #pragma unroll
    for (int n = 0; n < 2; ++n)
        #pragma unroll
        for (int e = 0; e < 8; ++e) acc[n][e] = 0.0f;

    auto stage = [&](int buf, int kt) {
        const float* srcA = w5 + (size_t)(tileM + aRow) * D + kt + aK8;
        const float* srcB = ht + (size_t)(kt + bK) * D + tileN + bN8;
        v4f ta0 = *(const v4f*)(srcA);
        v4f ta1 = *(const v4f*)(srcA + 4);
        v4f tb0 = *(const v4f*)(srcB);
        v4f tb1 = *(const v4f*)(srcB + 4);
        *(v8bf*)&As[buf][aRow][aK8] = cvt8r(ta0, ta1);
        v8bf bv = cvt8r(tb0, tb1);
        #pragma unroll
        for (int j = 0; j < 8; ++j) Bs[buf][bN8 + j][bK] = bv[j];
    };

    auto compute = [&](int buf) {
        v16bf aFrag = load_a_frag(&As[buf][0][0], waveM * 16 + lr, lhi);
        #pragma unroll
        for (int n = 0; n < 2; ++n) {
            v16bf bFrag = load_b_frag(&Bs[buf][0][0], waveN * 32 + n * 16 + lr, lhi);
            acc[n] = __builtin_amdgcn_wmma_f32_16x16x32_bf16(
                false, aFrag, false, bFrag, (short)0, acc[n], false, false);
        }
    };

    const int rowBase = tileM + waveM * 16 + lhi * 8;
    const int colBase = tileN + waveN * 32 + lr;

    stage(0, 0);
    int buf = 0;
    for (int kt = 0; kt < D; kt += 32) {
        __syncthreads();
        compute(buf);
        if (kt + 32 < D) {
            stage(buf ^ 1, kt + 32);
        } else {
            __builtin_prefetch(b5 + (size_t)rowBase * D + colBase, 0, 1);
        }
        buf ^= 1;
    }

    #pragma unroll
    for (int n = 0; n < 2; ++n) {
        const int col = colBase + n * 16;
        #pragma unroll
        for (int r = 0; r < 8; ++r) {
            const size_t idx = (size_t)(rowBase + r) * D + col;
            logits[idx] = acc[n][r] + b5[idx];
        }
    }
}

// ---------------------------------------------------------------------------
// Kernel 3: row-wise softmax, in place (one block per row)
// ---------------------------------------------------------------------------
__global__ __launch_bounds__(256)
void softmax_rows_kernel(float* __restrict__ y)
{
    __shared__ float red[256];
    float* p = y + (size_t)blockIdx.x * D;
    const int tid = threadIdx.x;

    float m = -1e30f;
    for (int c = tid; c < D; c += 256) m = fmaxf(m, p[c]);
    red[tid] = m; __syncthreads();
    for (int s = 128; s > 0; s >>= 1) {
        if (tid < s) red[tid] = fmaxf(red[tid], red[tid + s]);
        __syncthreads();
    }
    m = red[0]; __syncthreads();

    float sum = 0.0f;
    for (int c = tid; c < D; c += 256) {
        const float e = __expf(p[c] - m);
        p[c] = e;
        sum += e;
    }
    red[tid] = sum; __syncthreads();
    for (int s = 128; s > 0; s >>= 1) {
        if (tid < s) red[tid] += red[tid + s];
        __syncthreads();
    }
    const float inv = 1.0f / red[0];
    for (int c = tid; c < D; c += 256) p[c] *= inv;
}

// ---------------------------------------------------------------------------
extern "C" void kernel_launch(void* const* d_in, const int* in_sizes, int n_in,
                              void* d_out, int out_size, void* d_ws, size_t ws_size,
                              hipStream_t stream) {
    (void)in_sizes; (void)n_in; (void)out_size; (void)d_ws; (void)ws_size;

    const float* cPrev = (const float*)d_in[0];
    const float* hPrev = (const float*)d_in[1];
    const float* xt    = (const float*)d_in[2];
    const float* w1 = (const float*)d_in[3];  const float* b1 = (const float*)d_in[4];
    const float* w2 = (const float*)d_in[5];  const float* b2 = (const float*)d_in[6];
    const float* w3 = (const float*)d_in[7];  const float* b3 = (const float*)d_in[8];
    const float* w4 = (const float*)d_in[9];  const float* b4 = (const float*)d_in[10];
    const float* w5 = (const float*)d_in[11]; const float* b5 = (const float*)d_in[12];

    float* out = (float*)d_out;
    float* ct = out;                           // (D,D)
    float* ht = out + (size_t)D * D;           // (D,D)
    float* yt = out + (size_t)2 * D * D;       // (D,D)

    dim3 grid(D / 64, D / 64);
    dim3 block(256);

    lstm_gates_kernel<<<grid, block, 0, stream>>>(cPrev, hPrev, xt,
                                                  w1, b1, w2, b2, w3, b3, w4, b4,
                                                  ct, ht);
    lstm_out_gemm_kernel<<<grid, block, 0, stream>>>(w5, b5, ht, yt);
    softmax_rows_kernel<<<D, block, 0, stream>>>(yt);
}
